// MixedIntQuantizedLinear_41781441856135
// MI455X (gfx1250) — 8-bit matrix operand order repaired and checked by reading, compiled, not run on hardware
//
#include <hip/hip_runtime.h>
#include <stdint.h>


typedef int v8i __attribute__((ext_vector_type(8)));

#define IN_F   4096
#define OUT_F  4096
#define BLK    64
#define NBLK   (IN_F / BLK)
#define TOKENS 8192
#define BM     128
#define BN     128
#define PITCH  80
#define MT     2
#define NT     4

__global__ void pack_w_kernel(const int* __restrict__ W_int, uint8_t* __restrict__ W8) {
  int idx = blockIdx.x * blockDim.x + threadIdx.x;
  const int4* Wi = (const int4*)W_int;
  int4 v = Wi[idx];
  uint32_t p = ((uint32_t)v.x & 0xffu)
             | (((uint32_t)v.y & 0xffu) << 8)
             | (((uint32_t)v.z & 0xffu) << 16)
             | (((uint32_t)v.w & 0xffu) << 24);
  ((uint32_t*)W8)[idx] = p;
}

__global__ void quant_x_kernel(const float* __restrict__ x,
                               uint8_t* __restrict__ xq,
                               float* __restrict__ tscale) {
  __shared__ float red[8];
  __shared__ float s_scale;
  const int token = blockIdx.x;
  const float* xr = x + (size_t)token * IN_F;

  float m = 0.f;
  for (int i = threadIdx.x; i < IN_F; i += 256)
    m = fmaxf(m, fabsf(xr[i]));
  #pragma unroll
  for (int off = 16; off > 0; off >>= 1)
    m = fmaxf(m, __shfl_xor(m, off, 32));
  if ((threadIdx.x & 31) == 0) red[threadIdx.x >> 5] = m;
  __syncthreads();
  if (threadIdx.x == 0) {
    float mm = red[0];
    #pragma unroll
    for (int i = 1; i < 8; ++i) mm = fmaxf(mm, red[i]);
    mm = fmaxf(mm, 1e-8f);
    float s = mm * (1.0f / 127.0f);
    s_scale = s;
    tscale[token] = s;
  }
  __syncthreads();
  const float inv = 1.0f / s_scale;
  for (int i = threadIdx.x; i < IN_F; i += 256) {
    float q = rintf(xr[i] * inv);
    q = fminf(fmaxf(q, -127.0f), 127.0f);
    xq[(size_t)token * IN_F + i] = (uint8_t)(int8_t)(int)q;
  }
}

__global__ void __launch_bounds__(256)
gemm_kernel(const uint8_t* __restrict__ xq, const uint8_t* __restrict__ W8,
            const int* __restrict__ scale_i32, const float* __restrict__ bias,
            const float* __restrict__ tscale, float* __restrict__ out) {
  __shared__ __align__(16) uint8_t sA[2][BM * PITCH];
  __shared__ __align__(16) uint8_t sB[2][BN * PITCH];

  const int t     = threadIdx.x;
  const int lane  = t & 31;
  const int wave  = t >> 5;
  const int half  = lane >> 4;
  const int l16   = lane & 15;
  const int waveM = wave >> 1;
  const int waveN = wave & 1;

  const int m0 = blockIdx.x * BM;
  const int n0 = blockIdx.y * BN;

  const int rS = t >> 2;
  const int cS = (t & 3) * 16;
  const uint8_t* pA0 = xq + (size_t)(m0 + rS)      * IN_F + cS;
  const uint8_t* pA1 = xq + (size_t)(m0 + rS + 64) * IN_F + cS;
  const uint8_t* pB0 = W8 + (size_t)(n0 + rS)      * IN_F + cS;
  const uint8_t* pB1 = W8 + (size_t)(n0 + rS + 64) * IN_F + cS;
  const int dS0 = rS * PITCH + cS;
  const int dS1 = dS0 + 64 * PITCH;

  int aOff[MT], bOff[NT];
  const int* sPtr[NT];
  #pragma unroll
  for (int mt = 0; mt < MT; ++mt)
    aOff[mt] = (waveM * 32 + mt * 16 + l16) * PITCH + half * 16;
  #pragma unroll
  for (int nt = 0; nt < NT; ++nt) {
    const int rowB = waveN * 64 + nt * 16 + l16;
    bOff[nt] = rowB * PITCH + half * 16;
    sPtr[nt] = scale_i32 + (size_t)(n0 + rowB) * NBLK;
  }

  float acc[MT][NT][8];
  #pragma unroll
  for (int mt = 0; mt < MT; ++mt)
    #pragma unroll
    for (int nt = 0; nt < NT; ++nt)
      #pragma unroll
      for (int i = 0; i < 8; ++i) acc[mt][nt][i] = 0.f;

  const v8i zero = {0, 0, 0, 0, 0, 0, 0, 0};
  const float kInvShift = 1.0f / 16777216.0f;

  {
    uint4 a0 = *(const uint4*)pA0, a1 = *(const uint4*)pA1;
    uint4 b0 = *(const uint4*)pB0, b1 = *(const uint4*)pB1;
    pA0 += BLK; pA1 += BLK; pB0 += BLK; pB1 += BLK;
    *(uint4*)&sA[0][dS0] = a0;  *(uint4*)&sA[0][dS1] = a1;
    *(uint4*)&sB[0][dS0] = b0;  *(uint4*)&sB[0][dS1] = b1;
  }
  __syncthreads();

  for (int kb = 0; kb < NBLK; ++kb) {
    const int cur = kb & 1;

    uint4 na0, na1, nb0, nb1;
    if (kb < NBLK - 1) {
      na0 = *(const uint4*)pA0;  na1 = *(const uint4*)pA1;
      nb0 = *(const uint4*)pB0;  nb1 = *(const uint4*)pB1;
      pA0 += BLK; pA1 += BLK; pB0 += BLK; pB1 += BLK;
      if (kb < NBLK - 2) {
        __builtin_prefetch(pA0, 0, 1);
        __builtin_prefetch(pB0, 0, 1);
      }
    }

    v8i aF[MT], bF[NT];
    #pragma unroll
    for (int mt = 0; mt < MT; ++mt) {
      #pragma unroll
      for (int p = 0; p < 4; ++p) {
        uint2 w = *(const uint2*)&sA[cur][aOff[mt] + (p >> 1) * 32 + (p & 1) * 8];
        aF[mt][2 * p]     = (int)w.x;
        aF[mt][2 * p + 1] = (int)w.y;
      }
    }
    #pragma unroll
    for (int nt = 0; nt < NT; ++nt) {
      uint4 lo = *(const uint4*)&sB[cur][bOff[nt]];
      uint4 hi = *(const uint4*)&sB[cur][bOff[nt] + 32];
      bF[nt][0] = (int)lo.x; bF[nt][1] = (int)lo.y;
      bF[nt][2] = (int)lo.z; bF[nt][3] = (int)lo.w;
      bF[nt][4] = (int)hi.x; bF[nt][5] = (int)hi.y;
      bF[nt][6] = (int)hi.z; bF[nt][7] = (int)hi.w;
    }

    float sb[NT];
    #pragma unroll
    for (int nt = 0; nt < NT; ++nt)
      sb[nt] = (float)sPtr[nt][kb] * kInvShift;

    #pragma unroll
    for (int mt = 0; mt < MT; ++mt) {
      #pragma unroll
      for (int nt = 0; nt < NT; ++nt) {
        v8i d = __builtin_amdgcn_wmma_i32_16x16x64_iu8(
             true, aF[mt],  true, bF[nt], zero,
             false,  false);
        #pragma unroll
        for (int i = 0; i < 8; ++i)
          acc[mt][nt][i] += (float)d[i] * sb[nt];
      }
    }

    if (kb < NBLK - 1) {
      const int nxt = cur ^ 1;
      *(uint4*)&sA[nxt][dS0] = na0;  *(uint4*)&sA[nxt][dS1] = na1;
      *(uint4*)&sB[nxt][dS0] = nb0;  *(uint4*)&sB[nxt][dS1] = nb1;
      __syncthreads();
    }
  }

  #pragma unroll
  for (int nt = 0; nt < NT; ++nt) {
    const int n = n0 + waveN * 64 + nt * 16 + l16;
    const float bi = bias[n];
    #pragma unroll
    for (int mt = 0; mt < MT; ++mt) {
      #pragma unroll
      for (int i = 0; i < 8; ++i) {
        const int m = m0 + waveM * 32 + mt * 16 + i + 8 * half;
        out[(size_t)m * OUT_F + n] = acc[mt][nt][i] * tscale[m] + bi;
      }
    }
  }
}

extern "C" void kernel_launch(void* const* d_in, const int* in_sizes, int n_in,
                              void* d_out, int out_size, void* d_ws, size_t ws_size,
                              hipStream_t stream) {
  const float* x         = (const float*)d_in[0];
  const int*   W_int     = (const int*)d_in[1];
  const int*   scale_i32 = (const int*)d_in[2];
  const float* bias      = (const float*)d_in[3];
  float*       out       = (float*)d_out;

  uint8_t* ws = (uint8_t*)d_ws;
  uint8_t* W8     = ws;
  uint8_t* xqv    = ws + (size_t)OUT_F * IN_F;
  float*   tscale = (float*)(ws + (size_t)OUT_F * IN_F + (size_t)TOKENS * IN_F);

  pack_w_kernel<<<(OUT_F * IN_F / 4) / 256, 256, 0, stream>>>(W_int, W8);
  quant_x_kernel<<<TOKENS, 256, 0, stream>>>(x, xqv, tscale);

  dim3 grid(TOKENS / BM, OUT_F / BN);
  gemm_kernel<<<grid, 256, 0, stream>>>(xqv, W8, scale_i32, bias, tscale, out);
}
